// EGNNPosOnlyLayer_87625922772994
// MI455X (gfx1250) — matrix operand order repaired and checked by reading, not compiled, not run on hardware
//
#include <hip/hip_runtime.h>
#include <math.h>

typedef __attribute__((ext_vector_type(16))) _Float16 v16h;
typedef __attribute__((ext_vector_type(8)))  float    v8f;

constexpr int HD = 64;               // hidden feats
constexpr int WAVES_PER_BLOCK = 8;   // 256 threads = 8 wave32
constexpr int BLOCK = WAVES_PER_BLOCK * 32;

// Fast silu: x * rcp(1+exp(-x)) using v_rcp_f32 (approx) -- operands feed f16
// WMMA fragments, so approximate rcp is plenty.
__device__ __forceinline__ float silu_f(float x) {
    return x * __builtin_amdgcn_rcpf(1.0f + __expf(-x));
}

// DPP ROW_XMASK butterfly add: x += lane[x ^ MASK] within each 16-lane row.
// Pure VALU (v_mov_dpp + v_add), avoids the ds_bpermute LDS round-trip.
template <int MASK>
__device__ __forceinline__ float xadd_dpp(float x) {
    const int y = __builtin_amdgcn_update_dpp(0, __float_as_int(x),
                                              0x160 | MASK, 0xF, 0xF, true);
    return x + __int_as_float(y);
}

__device__ __forceinline__ float readlane_f(float x, int lane) {
    return __int_as_float(__builtin_amdgcn_readlane(__float_as_int(x), lane));
}

// A-matrix (16x32 f16) element->K map within a 32-wide K chunk.
// Lane l holds row M = l&15. Element e (0..15): VGPR v=e>>1, half h=e&1.
__device__ __forceinline__ int a_k(int lane, int e) {
    int v = e >> 1, hb = e & 1;
    int g      = v >> 2;            // 0 -> K base 0, 1 -> K base 16
    int within = v & 3;             // 2 K's per VGPR
    int khalf  = (lane >> 4) & 1;   // upper lane-half holds K+8
    return g * 16 + khalf * 8 + within * 2 + hb;
}

// B-matrix (32x16 f16) fragment from fp32 weight W[64][64] (row=k_in, col=n_out).
__device__ __forceinline__ v16h load_bfrag_f32(const float* __restrict__ W,
                                               int kc, int nt, int lane) {
    v16h b;
    const int n     = (lane & 15) + nt * 16;
    const int kbase = kc * 32 + ((lane >> 4) & 1) * 8;
#pragma unroll
    for (int e = 0; e < 16; ++e)
        b[e] = (_Float16)W[(kbase + (e < 8 ? e : e + 8)) * HD + n];
    return b;
}

__device__ __forceinline__ v8f wmma_f16(v16h a, v16h b, v8f c) {
    return __builtin_amdgcn_wmma_f32_16x16x32_f16(false, a, false, b,
                                                  (short)0, c, false, false);
}

// ---------------------------------------------------------------------------
// Kernel 0: zero agg, pos_out = pos
// ---------------------------------------------------------------------------
__global__ void egnn_init_kernel(const float* __restrict__ pos,
                                 float* __restrict__ pos_out,
                                 float* __restrict__ agg, int Nn) {
    const int stride = gridDim.x * blockDim.x;
    const int tid = blockIdx.x * blockDim.x + threadIdx.x;
    const int tot = Nn * HD;
    for (int i = tid; i < tot; i += stride) agg[i] = 0.0f;
    const int p3 = Nn * 3;
    for (int i = tid; i < p3; i += stride) pos_out[i] = pos[i];
}

// ---------------------------------------------------------------------------
// Kernel 1: fused edge pipeline.  One wave handles a tile of 16 edges:
//   dist -> layer1 (elementwise, straight into A fragments) -> WMMA layer2
//   -> silu -> atomic scatter into agg, and (e·cw)*rel/dist atomics into pos.
// ---------------------------------------------------------------------------
__global__ void __launch_bounds__(BLOCK)
egnn_edge_kernel(const float* __restrict__ pos, const int* __restrict__ ei,
                 const float* __restrict__ ew1, const float* __restrict__ eb1,
                 const float* __restrict__ ew2, const float* __restrict__ eb2,
                 const float* __restrict__ cw,
                 float* __restrict__ agg, float* __restrict__ pos_out, int E) {
    const int lane   = threadIdx.x & 31;
    const int wave   = threadIdx.x >> 5;
    const int gwave  = blockIdx.x * WAVES_PER_BLOCK + wave;
    const int nwaves = gridDim.x * WAVES_PER_BLOCK;
    const int ntiles = (E + 15) >> 4;
    const int hi = (lane >> 4) & 1;
    const int nl = lane & 15;

    // layer-1 params for exactly the (fragment-element -> K) slots this lane owns
    float w1k[32], b1k[32];
#pragma unroll
    for (int kc = 0; kc < 2; ++kc)
#pragma unroll
        for (int e = 0; e < 16; ++e) {
            const int k = kc * 32 + a_k(lane, e);
            w1k[kc * 16 + e] = ew1[k];
            b1k[kc * 16 + e] = eb1[k];
        }

    float cwn[4], eb2n[4];
#pragma unroll
    for (int nt = 0; nt < 4; ++nt) {
        cwn[nt]  = cw[nt * 16 + nl];
        eb2n[nt] = eb2[nt * 16 + nl];
    }

    // persistent B fragments of ew2: 4 N-tiles x 2 K-chunks
    v16h Bf[8];
#pragma unroll
    for (int nt = 0; nt < 4; ++nt)
#pragma unroll
        for (int kc = 0; kc < 2; ++kc)
            Bf[nt * 2 + kc] = load_bfrag_f32(ew2, kc, nt, lane);

    for (int tile = gwave; tile < ntiles; tile += nwaves) {
        const int edge     = tile * 16 + nl;       // both lane-halves mirror edges
        const bool fullTile = (tile * 16 + 16) <= E;   // uniform across the wave
        const bool valid   = edge < E;
        const int eC       = valid ? edge : (E - 1);
        const int row      = ei[eC];
        const int col      = ei[E + eC];
        const float rx = pos[row * 3 + 0] - pos[col * 3 + 0];
        const float ry = pos[row * 3 + 1] - pos[col * 3 + 1];
        const float rz = pos[row * 3 + 2] - pos[col * 3 + 2];
        const float dist = sqrtf(rx * rx + ry * ry + rz * rz);

        // layer 1 directly into A fragments (row M = nl is this lane's edge)
        v16h A0, A1;
#pragma unroll
        for (int e = 0; e < 16; ++e) {
            A0[e] = (_Float16)silu_f(dist * w1k[e]      + b1k[e]);
            A1[e] = (_Float16)silu_f(dist * w1k[16 + e] + b1k[16 + e]);
        }

        // destination rows for C layout (M = v + 8*hi) via readlane broadcasts
        int rowm[8];
#pragma unroll
        for (int v = 0; v < 8; ++v) {
            const int rlo = __builtin_amdgcn_readlane(row, v);      // row M=v
            const int rhi = __builtin_amdgcn_readlane(row, v + 8);  // row M=v+8
            rowm[v] = hi ? rhi : rlo;
        }

        // WMMA layer 2 + silu; keep values for offset-immediate atomics
        float e2v[4][8];
        float p[8];
#pragma unroll
        for (int v = 0; v < 8; ++v) p[v] = 0.0f;

#pragma unroll
        for (int nt = 0; nt < 4; ++nt) {
            v8f c;
#pragma unroll
            for (int v = 0; v < 8; ++v) c[v] = eb2n[nt];   // bias eb2[n]
            c = wmma_f16(A0, Bf[nt * 2 + 0], c);
            c = wmma_f16(A1, Bf[nt * 2 + 1], c);
#pragma unroll
            for (int v = 0; v < 8; ++v) {
                const float y = silu_f(c[v]);
                e2v[nt][v] = y;
                p[v] += y * cwn[nt];                        // partial e·cw
            }
        }

        // scatter-add e2 into agg: one base address per row, 4 atomics with
        // immediate offsets (0,64,128,192 bytes)
        if (fullTile) {
#pragma unroll
            for (int v = 0; v < 8; ++v) {
                float* base = agg + (size_t)rowm[v] * HD + nl;
#pragma unroll
                for (int nt = 0; nt < 4; ++nt)
                    atomicAdd(base + nt * 16, e2v[nt][v]);
            }
        } else {
#pragma unroll
            for (int v = 0; v < 8; ++v) {
                if (__shfl(valid ? 1 : 0, v + 8 * hi, 32)) {
                    float* base = agg + (size_t)rowm[v] * HD + nl;
#pragma unroll
                    for (int nt = 0; nt < 4; ++nt)
                        atomicAdd(base + nt * 16, e2v[nt][v]);
                }
            }
        }

        // reduce p[v] across the 16-lane row (DPP ROW_XMASK butterflies);
        // afterwards every lane of a row holds that row's full 64-wide dot.
#pragma unroll
        for (int v = 0; v < 8; ++v)
            p[v] = xadd_dpp<8>(xadd_dpp<4>(xadd_dpp<2>(xadd_dpp<1>(p[v]))));

        // route scalar to the edge-owner lane (m = nl, lanes 0..15):
        //   m < 8 : lane already holds it in p[m] (same row)
        //   m >= 8: row sum lives in the upper row -> readlane broadcast
        float s = 0.0f;
#pragma unroll
        for (int v = 0; v < 8; ++v) {
            if (nl == v) s = p[v];
            const float hs = readlane_f(p[v], 16);
            if (nl == v + 8) s = hs;
        }

        if (lane < 16 && valid) {
            const float inv = s * __builtin_amdgcn_rcpf(dist + 1e-8f);
            float* pb = pos_out + (size_t)row * 3;
            atomicAdd(pb + 0, rx * inv);
            atomicAdd(pb + 1, ry * inv);
            atomicAdd(pb + 2, rz * inv);
        }
    }
}

// ---------------------------------------------------------------------------
// Kernel 2: node MLP, h_out = silu(agg@nw1+nb1)@nw2 + nb2.
// C-layout -> A-layout transpose of the intermediate via per-wave LDS slice.
// ---------------------------------------------------------------------------
__global__ void __launch_bounds__(BLOCK)
egnn_node_kernel(const float* __restrict__ agg,
                 const float* __restrict__ nw1, const float* __restrict__ nb1,
                 const float* __restrict__ nw2, const float* __restrict__ nb2,
                 float* __restrict__ h_out, int Nn) {
    __shared__ _Float16 lds[WAVES_PER_BLOCK][16 * HD];
    const int lane   = threadIdx.x & 31;
    const int wave   = threadIdx.x >> 5;
    const int gwave  = blockIdx.x * WAVES_PER_BLOCK + wave;
    const int nwaves = gridDim.x * WAVES_PER_BLOCK;
    const int ntiles = (Nn + 15) >> 4;
    const int hi = (lane >> 4) & 1;
    const int nl = lane & 15;
    _Float16* myLds = lds[wave];

    float nb1n[4], nb2n[4];
#pragma unroll
    for (int nt = 0; nt < 4; ++nt) {
        nb1n[nt] = nb1[nt * 16 + nl];
        nb2n[nt] = nb2[nt * 16 + nl];
    }

    v16h B1[8], B2[8];
#pragma unroll
    for (int nt = 0; nt < 4; ++nt)
#pragma unroll
        for (int kc = 0; kc < 2; ++kc) {
            B1[nt * 2 + kc] = load_bfrag_f32(nw1, kc, nt, lane);
            B2[nt * 2 + kc] = load_bfrag_f32(nw2, kc, nt, lane);
        }

    for (int tile = gwave; tile < ntiles; tile += nwaves) {
        const int node  = tile * 16 + nl;               // A row this lane holds
        const int nodeC = node < Nn ? node : (Nn - 1);
        const float* aggRow = agg + (size_t)nodeC * HD;

        // A fragments: element pairs are K-consecutive -> vector float2 loads
        v16h A0, A1;
#pragma unroll
        for (int v = 0; v < 8; ++v) {
            const int k0 = a_k(lane, 2 * v);            // k for (e=2v, e=2v+1)
            const float2 f0 = *(const float2*)(aggRow + k0);
            const float2 f1 = *(const float2*)(aggRow + 32 + k0);
            A0[2 * v]     = (_Float16)f0.x;
            A0[2 * v + 1] = (_Float16)f0.y;
            A1[2 * v]     = (_Float16)f1.x;
            A1[2 * v + 1] = (_Float16)f1.y;
        }

        // GEMM 1 + silu, stash intermediate (f16) in LDS in row-major tile form
#pragma unroll
        for (int nt = 0; nt < 4; ++nt) {
            v8f c;
#pragma unroll
            for (int v = 0; v < 8; ++v) c[v] = nb1n[nt];
            c = wmma_f16(A0, B1[nt * 2 + 0], c);
            c = wmma_f16(A1, B1[nt * 2 + 1], c);
#pragma unroll
            for (int v = 0; v < 8; ++v) {
                const int m = v + 8 * hi;
                myLds[m * HD + nt * 16 + nl] = (_Float16)silu_f(c[v]);
            }
        }

        asm volatile("s_wait_dscnt 0" ::: "memory");     // DS stores visible

        v16h T0, T1;
#pragma unroll
        for (int e = 0; e < 16; ++e) {
            T0[e] = myLds[nl * HD +      a_k(lane, e)];
            T1[e] = myLds[nl * HD + 32 + a_k(lane, e)];
        }

        // GEMM 2 + bias, store fp32 result
#pragma unroll
        for (int nt = 0; nt < 4; ++nt) {
            v8f c;
#pragma unroll
            for (int v = 0; v < 8; ++v) c[v] = nb2n[nt];
            c = wmma_f16(T0, B2[nt * 2 + 0], c);
            c = wmma_f16(T1, B2[nt * 2 + 1], c);
#pragma unroll
            for (int v = 0; v < 8; ++v) {
                const int m = v + 8 * hi;
                const int nOut = tile * 16 + m;
                if (nOut < Nn)
                    h_out[(size_t)nOut * HD + nt * 16 + nl] = c[v];
            }
        }
        asm volatile("s_wait_dscnt 0" ::: "memory");     // reads done before reuse
    }
}

// ---------------------------------------------------------------------------
extern "C" void kernel_launch(void* const* d_in, const int* in_sizes, int n_in,
                              void* d_out, int out_size, void* d_ws, size_t ws_size,
                              hipStream_t stream) {
    // inputs (setup_inputs order): h, pos, edge_index, ew1, eb1, ew2, eb2, cw,
    //                              nw1, nb1, nw2, nb2   (h is unused by the math)
    const float* pos = (const float*)d_in[1];
    const int*   ei  = (const int*)d_in[2];
    const float* ew1 = (const float*)d_in[3];
    const float* eb1 = (const float*)d_in[4];
    const float* ew2 = (const float*)d_in[5];
    const float* eb2 = (const float*)d_in[6];
    const float* cw  = (const float*)d_in[7];
    const float* nw1 = (const float*)d_in[8];
    const float* nb1 = (const float*)d_in[9];
    const float* nw2 = (const float*)d_in[10];
    const float* nb2 = (const float*)d_in[11];

    const int Nn = in_sizes[1] / 3;   // 100,000
    const int E  = in_sizes[2] / 2;   // 3,200,000

    float* h_out   = (float*)d_out;                       // [N, 64]
    float* pos_out = (float*)d_out + (size_t)Nn * HD;     // [N, 3]
    float* agg     = (float*)d_ws;                        // [N, 64] scratch

    egnn_init_kernel<<<512, 256, 0, stream>>>(pos, pos_out, agg, Nn);

    const int etiles  = (E + 15) / 16;
    int eblocks = (etiles + WAVES_PER_BLOCK - 1) / WAVES_PER_BLOCK;
    if (eblocks > 1024) eblocks = 1024;                   // 8192 waves, grid-stride
    egnn_edge_kernel<<<eblocks, BLOCK, 0, stream>>>(pos, ei, ew1, eb1, ew2, eb2,
                                                    cw, agg, pos_out, E);

    const int ntiles  = (Nn + 15) / 16;
    const int nblocks = (ntiles + WAVES_PER_BLOCK - 1) / WAVES_PER_BLOCK;
    egnn_node_kernel<<<nblocks, BLOCK, 0, stream>>>(agg, nw1, nb1, nw2, nb2,
                                                    h_out, Nn);
}
